// BertAlibiUnpadSelfAttention_12747462935121
// MI455X (gfx1250) — compile-verified, matrix operand order checked by reading, not run on hardware
//
#include <hip/hip_runtime.h>
#include <hip/hip_bf16.h>
#include <math.h>

// ---------------------------------------------------------------------------
// BERT ALiBi unpadded self-attention for MI455X (gfx1250, wave32, WMMA).
// Roofline: bias tensor (B*H*S*S*4B = 402MB) dominates HBM traffic (23.3TB/s
// -> ~17us floor) => bias loads are non-temporal + prefetched. Q/K/V in f16
// (12.6MB each) fit in the 192MB L2 and are re-read from there. All matmuls
// use v_wmma_f32_16x16x32_f16. GEMM tiles are staged into LDS with the CDNA5
// async path (global_load_async_to_lds_b128 / s_wait_asynccnt).
// ---------------------------------------------------------------------------

typedef _Float16 h8   __attribute__((ext_vector_type(8)));
typedef _Float16 v16h __attribute__((ext_vector_type(16)));
typedef float    v8f  __attribute__((ext_vector_type(8)));
typedef float    f4v  __attribute__((ext_vector_type(4)));

#define H_    12
#define DIM_  768
#define HD_   64
#define B_    8
#define S_    1024
#define NNZ_  (B_ * S_)
#define QKVN_ (3 * DIM_)   // 2304

static __device__ __forceinline__ v16h make_v16(h8 lo, h8 hi) {
  return __builtin_shufflevector(lo, hi, 0,1,2,3,4,5,6,7,8,9,10,11,12,13,14,15);
}
static __device__ __forceinline__ v8f wmma_f16(v16h a, v16h b, v8f c) {
  // D(16x16,f32) = A(16x32,f16) * B(32x16,f16) + C
  return __builtin_amdgcn_wmma_f32_16x16x32_f16(false, a, false, b, (short)0, c,
                                                false, false);
}

// CDNA5 async memory->LDS copy (ASYNCcnt-tracked); vdst = LDS byte offset.
static __device__ __forceinline__ void async_copy_b128(unsigned lds_off,
                                                       const _Float16* g) {
  asm volatile("global_load_async_to_lds_b128 %0, %1, off"
               :: "v"(lds_off), "v"((unsigned long long)(uintptr_t)g)
               : "memory");
}
static __device__ __forceinline__ void wait_async0() {
  asm volatile("s_wait_asynccnt 0x0" ::: "memory");
}
static __device__ __forceinline__ unsigned lds_off32(const void* p) {
  // generic LDS pointer: addr[31:0] is the LDS byte offset (ISA 10.2 aperture)
  return (unsigned)(uintptr_t)p;
}

// ---------------------------------------------------------------------------
// Kernel 0a: zero f16 Q/K/Vt workspace (generic-safe for sparse indices)
// ---------------------------------------------------------------------------
__global__ void zero_kernel(uint4* __restrict__ p, long n16) {
  long i = (long)blockIdx.x * blockDim.x + threadIdx.x;
  if (i < n16) {
    uint4 z; z.x = 0u; z.y = 0u; z.z = 0u; z.w = 0u;
    p[i] = z;
  }
}

// ---------------------------------------------------------------------------
// Kernel 0b: one-shot f32 -> f16 convert (hidden, W) so the GEMM K-loop can
// use pure async LDS staging with no per-iteration cvt VALU work.
// ---------------------------------------------------------------------------
__global__ void cvt_f16_kernel(const float4* __restrict__ src,
                               _Float16* __restrict__ dst, long n4) {
  long i = (long)blockIdx.x * blockDim.x + threadIdx.x;
  if (i >= n4) return;
  float4 v = src[i];
  _Float16* d = dst + i * 4;
  d[0] = (_Float16)v.x; d[1] = (_Float16)v.y;
  d[2] = (_Float16)v.z; d[3] = (_Float16)v.w;
}

// ---------------------------------------------------------------------------
// Kernel 1: QKV GEMM  (NNZ x 768) @ (768 x 2304) + b, scatter to Q/K/Vt (f16)
// Block tile 128(M) x 128(N), K-step 32. 8 waves, each computes 64x32.
// LDS row stride 56 halves (112B): 16B aligned + bank-conflict-free b128 reads.
// A/B tiles staged memory->LDS with global_load_async_to_lds_b128.
// ---------------------------------------------------------------------------
#define K1_KS 56

__global__ __launch_bounds__(256) void qkv_gemm_kernel(
    const _Float16* __restrict__ hiddenH, const _Float16* __restrict__ WH,
    const float* __restrict__ bvec,       const int*   __restrict__ indices,
    _Float16* __restrict__ Qh, _Float16* __restrict__ Kh,
    _Float16* __restrict__ Vt)
{
  __shared__ _Float16 Ah[128 * K1_KS];   // [m][k]
  __shared__ _Float16 Bh[128 * K1_KS];   // [n][k]  (W is row-major (n,k))
  __shared__ int idxS[128];

  const int tid   = threadIdx.x;
  const int lane  = tid & 31;
  const int wave  = tid >> 5;
  const int mTile = blockIdx.x % (NNZ_ / 128);   // 64
  const int nTile = blockIdx.x / (NNZ_ / 128);   // 18

  if (tid < 128) idxS[tid] = indices[mTile * 128 + tid];

  const int wm    = wave & 1;    // M half (64 rows)
  const int wn    = wave >> 1;   // N quarter (32 cols)
  const int m16   = lane & 15;
  const int khalf = lane >> 4;

  const v8f vzero = {};
  v8f acc[4][2];
  for (int i = 0; i < 4; ++i)
    for (int j = 0; j < 2; ++j) acc[i][j] = vzero;

  const int r  = tid >> 1;       // 0..127 (tile row this thread stages)
  const int hh = tid & 1;        // which 16-half chunk of the 32-wide K slice
  const _Float16* gA = hiddenH + (size_t)(mTile * 128 + r) * DIM_ + hh * 16;
  const _Float16* gB = WH      + (size_t)(nTile * 128 + r) * DIM_ + hh * 16;
  const unsigned ldsA = lds_off32(&Ah[r * K1_KS + hh * 16]);
  const unsigned ldsB = lds_off32(&Bh[r * K1_KS + hh * 16]);

  for (int k0 = 0; k0 < DIM_; k0 += 32) {
    __syncthreads();                       // all waves done reading prev tiles
    async_copy_b128(ldsA,      gA + k0);
    async_copy_b128(ldsA + 16, gA + k0 + 8);
    async_copy_b128(ldsB,      gB + k0);
    async_copy_b128(ldsB + 16, gB + k0 + 8);
    wait_async0();                         // this wave's DMA landed
    __syncthreads();                       // everyone's DMA landed

    // A frags: lane = row m16, K chunks [8*khalf..+7] and [16+8*khalf..+7]
    v16h af[4], bf[2];
    for (int mi = 0; mi < 4; ++mi) {
      const _Float16* base = &Ah[(wm * 64 + mi * 16 + m16) * K1_KS];
      af[mi] = make_v16(*(const h8*)(base + khalf * 8),
                        *(const h8*)(base + 16 + khalf * 8));
    }
    for (int ni = 0; ni < 2; ++ni) {
      const _Float16* base = &Bh[(wn * 32 + ni * 16 + m16) * K1_KS];
      bf[ni] = make_v16(*(const h8*)(base + khalf * 8),
                        *(const h8*)(base + 16 + khalf * 8));
    }
    for (int mi = 0; mi < 4; ++mi)
      for (int ni = 0; ni < 2; ++ni)
        acc[mi][ni] = wmma_f16(af[mi], bf[ni], acc[mi][ni]);
  }

  // Epilogue: + bias, scatter via indices into Q (pre-scaled by 1/sqrt(64)),
  // K [bh][s][d], and transposed Vt [bh][d][s].
  for (int ni = 0; ni < 2; ++ni) {
    const int gcol = nTile * 128 + wn * 32 + ni * 16 + m16;
    const float bval = bvec[gcol];
    const int sect = gcol / DIM_;         // 0=q, 1=k, 2=v
    const int rem  = gcol % DIM_;
    const int h    = rem / HD_;
    const int d    = rem % HD_;
    for (int mi = 0; mi < 4; ++mi) {
      for (int rr = 0; rr < 8; ++rr) {
        const int lrow = wm * 64 + mi * 16 + rr + 8 * khalf;  // C-layout row
        const int p = idxS[lrow];                             // padded row
        if (p < 0 || p >= B_ * S_) continue;
        const int bb = p >> 10;           // /S_
        const int ss = p & (S_ - 1);
        const size_t bh = (size_t)(bb * H_ + h);
        const float val = acc[mi][ni][rr] + bval;
        if (sect == 0)
          Qh[(bh * S_ + ss) * HD_ + d] = (_Float16)(val * 0.125f); // 1/sqrt(64)
        else if (sect == 1)
          Kh[(bh * S_ + ss) * HD_ + d] = (_Float16)val;
        else
          Vt[(bh * HD_ + d) * S_ + ss] = (_Float16)val;
      }
    }
  }
}

// ---------------------------------------------------------------------------
// Kernel 2: flash attention. One wave per 16 q rows; 8 waves/block => 128
// q rows per block. Loop over 32 keys/iter: 4 WMMAs for scores, NT bias add,
// online softmax (shfl_xor reductions over 16-lane halves), probs->LDS
// transpose to A-layout, 4 WMMAs into the 16x64 f32 output accumulator.
// ---------------------------------------------------------------------------
#define P_KS 48   // probs LDS row stride in halves (96B: 16B aligned)

__global__ __launch_bounds__(256) void attn_kernel(
    const _Float16* __restrict__ Qh, const _Float16* __restrict__ Kh,
    const _Float16* __restrict__ Vt, const float* __restrict__ bias,
    float* __restrict__ attnp)
{
  __shared__ _Float16 plds[8][16 * P_KS];   // per-wave probs scratch

  const int tid  = threadIdx.x;
  const int lane = tid & 31;
  const int wave = tid >> 5;
  const int nqb  = S_ / 128;                 // 8 q-blocks per (b,h)
  const int qblk = blockIdx.x % nqb;
  const int bh   = blockIdx.x / nqb;         // 0..B*H-1
  const int h    = bh % H_;
  const int b    = bh / H_;
  const int q0   = qblk * 128 + wave * 16;
  const int m16  = lane & 15;
  const int khalf = lane >> 4;

  const _Float16* Qb = Qh + (size_t)bh * S_ * HD_;
  const _Float16* Kb = Kh + (size_t)bh * S_ * HD_;
  const _Float16* Vb = Vt + (size_t)bh * HD_ * S_;
  const float* biasB = bias + (size_t)bh * S_ * S_;

  // Q fragments: 16 rows x 64 d, split into two K=32 frags (q pre-scaled).
  v16h qf[2];
  for (int f = 0; f < 2; ++f) {
    const _Float16* p = Qb + (size_t)(q0 + m16) * HD_ + f * 32;
    qf[f] = make_v16(*(const h8*)(p + khalf * 8),
                     *(const h8*)(p + 16 + khalf * 8));
  }

  const v8f vzero = {};
  v8f out[4];
  for (int dt = 0; dt < 4; ++dt) out[dt] = vzero;
  float rm[8], rs[8];
  for (int rr = 0; rr < 8; ++rr) { rm[rr] = -1.0e30f; rs[rr] = 0.0f; }

  _Float16* pw = &plds[wave][0];

  for (int kt = 0; kt < S_ / 32; ++kt) {
    const int kbase = kt * 32;
    const float* bp = biasB + (size_t)(q0 + 8 * khalf) * S_ + kbase + m16;

    // prefetch next bias tile (streamed once: NT path, roofline term)
    if (kt + 1 < S_ / 32) {
      for (int rr = 0; rr < 8; ++rr)
        __builtin_prefetch(bp + (size_t)rr * S_ + 32, 0, 0);
    }

    // scores: 16 q x 32 keys = two 16x16 accum tiles, K over d=64
    v8f s[2]; s[0] = vzero; s[1] = vzero;
    for (int nh = 0; nh < 2; ++nh) {
      const _Float16* kp = Kb + (size_t)(kbase + nh * 16 + m16) * HD_;
      for (int f = 0; f < 2; ++f) {
        v16h bf = make_v16(*(const h8*)(kp + f * 32 + khalf * 8),
                           *(const h8*)(kp + f * 32 + 16 + khalf * 8));
        s[nh] = wmma_f16(qf[f], bf, s[nh]);
      }
    }

    // + bias: non-temporal so the 402MB stream does not evict Q/K/V from L2
    for (int nh = 0; nh < 2; ++nh)
      for (int rr = 0; rr < 8; ++rr)
        s[nh][rr] += __builtin_nontemporal_load(bp + (size_t)rr * S_ + nh * 16);

    // online softmax: row max over 32 keys (16-lane half holds one row/VGPR)
    float scale[8];
    for (int rr = 0; rr < 8; ++rr) {
      float v = fmaxf(s[0][rr], s[1][rr]);
      v = fmaxf(v, __shfl_xor(v, 1));
      v = fmaxf(v, __shfl_xor(v, 2));
      v = fmaxf(v, __shfl_xor(v, 4));
      v = fmaxf(v, __shfl_xor(v, 8));
      const float nm = fmaxf(rm[rr], v);
      scale[rr] = __expf(rm[rr] - nm);
      rm[rr] = nm;
    }
    for (int rr = 0; rr < 8; ++rr) {
      const float p0 = __expf(s[0][rr] - rm[rr]);
      const float p1 = __expf(s[1][rr] - rm[rr]);
      s[0][rr] = p0; s[1][rr] = p1;
      float ps = p0 + p1;
      ps += __shfl_xor(ps, 1);
      ps += __shfl_xor(ps, 2);
      ps += __shfl_xor(ps, 4);
      ps += __shfl_xor(ps, 8);
      rs[rr] = rs[rr] * scale[rr] + ps;
      for (int dt = 0; dt < 4; ++dt) out[dt][rr] *= scale[rr];
    }

    // probs (C layout) -> LDS -> A layout (same-wave DS ordering suffices)
    for (int nh = 0; nh < 2; ++nh)
      for (int rr = 0; rr < 8; ++rr)
        pw[(rr + 8 * khalf) * P_KS + nh * 16 + m16] = (_Float16)s[nh][rr];
    v16h pa = make_v16(*(const h8*)(pw + m16 * P_KS + khalf * 8),
                       *(const h8*)(pw + m16 * P_KS + 16 + khalf * 8));

    // out += probs(16x32) @ V(32x64); Vt transposed so chunks are contiguous
    for (int dt = 0; dt < 4; ++dt) {
      const _Float16* vp = Vb + (size_t)(dt * 16 + m16) * S_ + kbase;
      v16h bf = make_v16(*(const h8*)(vp + khalf * 8),
                         *(const h8*)(vp + 16 + khalf * 8));
      out[dt] = wmma_f16(pa, bf, out[dt]);
    }
  }

  // normalize and write padded attention rows (f32; re-read by gather => RT)
  for (int dt = 0; dt < 4; ++dt) {
    for (int rr = 0; rr < 8; ++rr) {
      const int qrow = q0 + rr + 8 * khalf;
      const float val = out[dt][rr] / rs[rr];
      attnp[(size_t)(b * S_ + qrow) * DIM_ + h * HD_ + dt * 16 + m16] = val;
    }
  }
}

// ---------------------------------------------------------------------------
// Kernel 3: final row gather  out[i] = attnp[indices[i]]  (NT stores: the
// result is never re-read on device). Uses clang ext-vector float4 so the
// nontemporal builtins accept the pointer type.
// ---------------------------------------------------------------------------
__global__ void gather_kernel(const f4v* __restrict__ attnp,
                              const int* __restrict__ indices,
                              f4v* __restrict__ out)
{
  const int F4 = DIM_ / 4;   // 192
  long i = (long)blockIdx.x * blockDim.x + threadIdx.x;
  if (i >= (long)NNZ_ * F4) return;
  const int row = (int)(i / F4);
  const int c   = (int)(i % F4);
  f4v v = attnp[(long)indices[row] * F4 + c];
  __builtin_nontemporal_store(v, &out[i]);
}

// ---------------------------------------------------------------------------
extern "C" void kernel_launch(void* const* d_in, const int* in_sizes, int n_in,
                              void* d_out, int out_size, void* d_ws,
                              size_t ws_size, hipStream_t stream)
{
  (void)in_sizes; (void)n_in; (void)out_size; (void)ws_size;
  const float* hidden  = (const float*)d_in[0];
  const int*   indices = (const int*)d_in[3];
  const float* bias    = (const float*)d_in[5];
  const float* W       = (const float*)d_in[7];
  const float* bvec    = (const float*)d_in[8];

  char* ws = (char*)d_ws;
  const size_t QB = (size_t)B_ * H_ * S_ * HD_ * sizeof(_Float16); // 12.58MB
  const size_t ATTNB = (size_t)B_ * S_ * DIM_ * sizeof(float);     // 25.17MB
  const size_t HHB   = (size_t)NNZ_ * DIM_ * sizeof(_Float16);     // 12.58MB
  _Float16* Qh      = (_Float16*)(ws);
  _Float16* Kh      = (_Float16*)(ws + QB);
  _Float16* Vt      = (_Float16*)(ws + 2 * QB);
  float*    attnp   = (float*)(ws + 3 * QB);
  _Float16* hiddenH = (_Float16*)(ws + 3 * QB + ATTNB);
  _Float16* WH      = (_Float16*)(ws + 3 * QB + ATTNB + HHB);

  const long zeroN = (long)(3 * QB) / 16;
  zero_kernel<<<(unsigned)((zeroN + 255) / 256), 256, 0, stream>>>(
      (uint4*)ws, zeroN);

  const long nh4 = (long)NNZ_ * DIM_ / 4;
  cvt_f16_kernel<<<(unsigned)((nh4 + 255) / 256), 256, 0, stream>>>(
      (const float4*)hidden, hiddenH, nh4);
  const long nw4 = (long)QKVN_ * DIM_ / 4;
  cvt_f16_kernel<<<(unsigned)((nw4 + 255) / 256), 256, 0, stream>>>(
      (const float4*)W, WH, nw4);

  qkv_gemm_kernel<<<(NNZ_ / 128) * (QKVN_ / 128), 256, 0, stream>>>(
      hiddenH, WH, bvec, indices, Qh, Kh, Vt);

  attn_kernel<<<B_ * H_ * (S_ / 128), 256, 0, stream>>>(
      Qh, Kh, Vt, bias, attnp);

  const long gN = (long)NNZ_ * (DIM_ / 4);
  gather_kernel<<<(unsigned)((gN + 255) / 256), 256, 0, stream>>>(
      (const f4v*)attnp, indices, (f4v*)d_out);
}
